// GNNActorVariable_27573690040351
// MI455X (gfx1250) — matrix operand order repaired and checked by reading, not compiled, not run on hardware
//
#include <hip/hip_runtime.h>
#include <hip/hip_bf16.h>
#include <math.h>

static constexpr int Bn   = 32;
static constexpr int Nn   = 1024;
static constexpr int Dd   = 64;
static constexpr int Hh   = 128;
static constexpr int OUTD = 8;
static constexpr int KNN  = 16;
static constexpr int MN   = Bn * Nn;   // 32768 nodes total

typedef __attribute__((ext_vector_type(16))) _Float16 v16h;
typedef __attribute__((ext_vector_type(8)))  _Float16 v8h;
typedef __attribute__((ext_vector_type(8)))  float    v8f;

// ---------------------------------------------------------------- mask
__global__ void mask_kernel(const float* __restrict__ obs, float* __restrict__ maskf) {
    int i = blockIdx.x * blockDim.x + threadIdx.x;
    if (i >= MN) return;
    const float4* row = (const float4*)(obs + (size_t)i * Dd);
    bool any = false;
#pragma unroll
    for (int d = 0; d < Dd / 4; ++d) {
        float4 v = row[d];
        any |= (v.x != 0.0f) | (v.y != 0.0f) | (v.z != 0.0f) | (v.w != 0.0f);
    }
    maskf[i] = any ? 1.0f : 0.0f;
}

// ---------------------------------------------------------------- kNN (top-17 by d2, LDS-staged positions)
__global__ __launch_bounds__(256) void knn_kernel(const float* __restrict__ obs,
                                                  const float* __restrict__ maskf,
                                                  int* __restrict__ nbr) {
    __shared__ float spx[Nn];
    __shared__ float spy[Nn];
    __shared__ float sm[Nn];
    int b     = blockIdx.x >> 2;   // 4 chunks of 256 nodes per batch
    int chunk = blockIdx.x & 3;
    int tid   = threadIdx.x;

    for (int j = tid; j < Nn; j += 256) {
        size_t base = (size_t)b * Nn + j;
        spx[j] = obs[base * Dd + 0];
        spy[j] = obs[base * Dd + 1];
        sm[j]  = maskf[base];
    }
    __syncthreads();

    int   il = chunk * 256 + tid;
    float xi = spx[il], yi = spy[il], mi = sm[il];

    float bd[KNN + 1];
    int   bi[KNN + 1];
#pragma unroll
    for (int p = 0; p <= KNN; ++p) { bd[p] = INFINITY; bi[p] = p; }

    for (int j = 0; j < Nn; ++j) {
        float dx = xi - spx[j];
        float dy = yi - spy[j];
        bool  valid = (mi != 0.0f) && (sm[j] != 0.0f);
        float d2 = valid ? (dx * dx + dy * dy) : INFINITY;
        if (d2 < bd[KNN]) {                 // register-resident insertion sort (stable: strict <)
            float cd = d2; int ci = j;
#pragma unroll
            for (int p = 0; p <= KNN; ++p) {
                if (cd < bd[p]) {
                    float td = bd[p]; int ti = bi[p];
                    bd[p] = cd; bi[p] = ci;
                    cd = td; ci = ti;
                }
            }
        }
    }
    size_t gi = (size_t)b * Nn + il;
#pragma unroll
    for (int k = 0; k < KNN; ++k) nbr[gi * KNN + k] = bi[k + 1];  // drop self
}

// ---------------------------------------------------------------- degree / dinv
__global__ void deg_init_kernel(const float* __restrict__ maskf, float* __restrict__ deg) {
    int i = blockIdx.x * blockDim.x + threadIdx.x;
    if (i < MN) deg[i] = 2.0f * maskf[i];
}

__global__ void deg_scatter_kernel(const float* __restrict__ maskf,
                                   const int* __restrict__ nbr,
                                   float* __restrict__ deg) {
    int i = blockIdx.x * blockDim.x + threadIdx.x;
    if (i >= MN) return;
    float w = maskf[i];
    if (w == 0.0f) return;
    int b = i >> 10;
#pragma unroll
    for (int k = 0; k < KNN; ++k) {
        int n = nbr[(size_t)i * KNN + k];
        atomicAdd(&deg[b * Nn + n], w);
    }
}

__global__ void dinv_kernel(float* __restrict__ deg) {
    int i = blockIdx.x * blockDim.x + threadIdx.x;
    if (i >= MN) return;
    float d = deg[i];
    deg[i] = (d > 0.0f) ? rsqrtf(d) : 0.0f;
}

// ---------------------------------------------------------------- f32 -> f16 row-major pack (for A matrices)
__global__ void packA_kernel(const float* __restrict__ src, _Float16* __restrict__ dst, int n) {
    int i = blockIdx.x * blockDim.x + threadIdx.x;   // over n/4 float4 groups
    if (i * 4 >= n) return;
    float4 v = ((const float4*)src)[i];
    v8h* d4 = nullptr; (void)d4;
    _Float16* d = dst + (size_t)i * 4;
    d[0] = (_Float16)v.x; d[1] = (_Float16)v.y; d[2] = (_Float16)v.z; d[3] = (_Float16)v.w;
}

// ---------------------------------------------------------------- W -> fragment-major f16 pack
// Fragment index t = kt*256 + ntile*32 + lane ; 16 contiguous halves per fragment:
//   b[e] = W[(kt*32 + (lane>=16?16:0) + e) * Hh + ntile*16 + (lane&15)]
template<int KD>
__global__ void packW_kernel(const float* __restrict__ W, _Float16* __restrict__ Wp) {
    int t = blockIdx.x * blockDim.x + threadIdx.x;
    constexpr int nFrag = (KD / 32) * 8 * 32;
    if (t >= nFrag) return;
    int lane  = t & 31;
    int ntile = (t >> 5) & 7;
    int kt    = t >> 8;
    int kb    = kt * 32 + ((lane >= 16) ? 16 : 0);
    int ncol  = ntile * 16 + (lane & 15);
    _Float16* out = Wp + (size_t)t * 16;
#pragma unroll
    for (int e = 0; e < 16; ++e) out[e] = (_Float16)W[(size_t)(kb + e) * Hh + ncol];
}

// ---------------------------------------------------------------- WMMA GEMM: Y(MN x 128) = A16(MN x KD) @ W(KD x 128)
// One m-tile (16 rows) per block; 8 waves cover the 8 n-tiles. A tile staged in LDS
// (padded rows -> no systematic bank aliasing). B fragments pre-swizzled in global.
template<int KD>
__global__ __launch_bounds__(256) void gemm_wmma_kernel(const _Float16* __restrict__ A16,
                                                        const _Float16* __restrict__ Wp,
                                                        float* __restrict__ Y) {
    constexpr int LDK = KD + 8;                       // +16B row pad
    __shared__ _Float16 sA[16 * LDK];

    const int lane = threadIdx.x & 31;
    const int wave = threadIdx.x >> 5;
    const int row0 = blockIdx.x * 16;

    // cooperative copy: 16 rows x KD halves, b128 granules
    constexpr int vecPerRow = KD / 8;
    constexpr int totalVec  = 16 * vecPerRow;
    for (int t = threadIdx.x; t < totalVec; t += 256) {
        int r = t / vecPerRow;
        int c = t % vecPerRow;
        *(v8h*)(sA + r * LDK + c * 8) =
            *(const v8h*)(A16 + (size_t)row0 * KD + (size_t)t * 8);
    }
    __syncthreads();

    const int ntile = wave;                            // 0..7
    const int khalf = (lane >= 16) ? 8 : 0;            // A lane-half K offset
    const int mrow  = lane & 15;
    const _Float16* aRow = sA + mrow * LDK;

    v8f acc = {};
#pragma unroll
    for (int kt = 0; kt < KD / 32; ++kt) {
        const int k0 = kt * 32;
        v8h  alo = *(const v8h*)(aRow + k0 + 2 * khalf);        // K = k0+2*khalf .. +7
        v8h  ahi = *(const v8h*)(aRow + k0 + 2 * khalf + 8);    // K = k0+2*khalf+8 .. +7
        v16h a   = __builtin_shufflevector(alo, ahi, 0,1,2,3,4,5,6,7,8,9,10,11,12,13,14,15);
        v16h b   = *(const v16h*)(Wp + ((size_t)(kt * 8 + ntile) * 32 + lane) * 16);
        acc = __builtin_amdgcn_wmma_f32_16x16x32_f16(
            /*neg_a=*/false, a, /*neg_b=*/false, b,
            /*c_mod=*/(short)0, acc, /*reuse_a=*/false, /*reuse_b=*/false);
    }

    // C/D layout: VGPR r -> M = row0 + r + (lane>=16 ? 8 : 0), N = ntile*16 + (lane&15)
    const int mbase = row0 + ((lane >= 16) ? 8 : 0);
    const int ncol  = ntile * 16 + (lane & 15);
#pragma unroll
    for (int r = 0; r < 8; ++r)
        Y[(size_t)(mbase + r) * Hh + ncol] = acc[r];
}

// ---------------------------------------------------------------- aggregation
__global__ void zero_kernel(float* __restrict__ p, int n) {
    int i = blockIdx.x * blockDim.x + threadIdx.x;
    if (i < n) p[i] = 0.0f;
}

__global__ __launch_bounds__(128) void agg_scatter_kernel(const float* __restrict__ y,
                                                          const float* __restrict__ dinv,
                                                          const float* __restrict__ maskf,
                                                          const int* __restrict__ nbr,
                                                          float* __restrict__ agg) {
    int i = blockIdx.x;       // node
    int c = threadIdx.x;      // channel
    float w = maskf[i];
    if (w == 0.0f) return;    // w in {0,1}; z = dinv*y*w
    float val = dinv[i] * y[(size_t)i * Hh + c];
    int b = i >> 10;
#pragma unroll
    for (int k = 0; k < KNN; ++k) {
        int n = nbr[(size_t)i * KNN + k];
        atomicAdd(&agg[(size_t)(b * Nn + n) * Hh + c], val);
    }
}

// combine + tanh; emits both f32 h (output head) and f16 h (next WMMA layer)
__global__ void combine_kernel(const float* __restrict__ y,
                               const float* __restrict__ agg,
                               const float* __restrict__ dinv,
                               const float* __restrict__ bias,
                               float* __restrict__ hout,
                               _Float16* __restrict__ hout16) {
    int t = blockIdx.x * blockDim.x + threadIdx.x;  // MN*Hh
    if (t >= MN * Hh) return;
    int i = t >> 7;
    int c = t & (Hh - 1);
    float di = dinv[i];
    float v  = di * agg[t] + 2.0f * di * di * y[t];
    float r  = tanhf(v + bias[c]);
    hout[t]   = r;
    hout16[t] = (_Float16)r;
}

// ---------------------------------------------------------------- output head (OUT=8 < 16, plain VALU)
__global__ void out_kernel(const float* __restrict__ h,
                           const float* __restrict__ Wout,
                           const float* __restrict__ bout,
                           const float* __restrict__ maskf,
                           float* __restrict__ out) {
    int t = blockIdx.x * blockDim.x + threadIdx.x;  // MN*OUTD
    if (t >= MN * OUTD) return;
    int i = t >> 3;
    int o = t & 7;
    float s = bout[o];
    const float* hr = h + (size_t)i * Hh;
#pragma unroll 8
    for (int c = 0; c < Hh; ++c) s += hr[c] * Wout[c * OUTD + o];
    out[t] = s * maskf[i];
}

// ---------------------------------------------------------------- launch
extern "C" void kernel_launch(void* const* d_in, const int* in_sizes, int n_in,
                              void* d_out, int out_size, void* d_ws, size_t ws_size,
                              hipStream_t stream) {
    (void)in_sizes; (void)n_in; (void)out_size; (void)ws_size;
    const float* obs  = (const float*)d_in[0];
    const float* W1   = (const float*)d_in[1];
    const float* b1   = (const float*)d_in[2];
    const float* W2   = (const float*)d_in[3];
    const float* b2   = (const float*)d_in[4];
    const float* Wout = (const float*)d_in[5];
    const float* bout = (const float*)d_in[6];
    float* out = (float*)d_out;

    char* ws = (char*)d_ws;
    size_t off = 0;
    auto alloc = [&](size_t bytes) -> void* {
        void* p = ws + off;
        off = (off + bytes + 255) & ~(size_t)255;
        return p;
    };
    float*    hbuf   = (float*)   alloc(sizeof(float)    * MN * Hh);   // 16 MB
    float*    ybuf   = (float*)   alloc(sizeof(float)    * MN * Hh);   // 16 MB
    float*    aggbuf = (float*)   alloc(sizeof(float)    * MN * Hh);   // 16 MB
    int*      nbr    = (int*)     alloc(sizeof(int)      * MN * KNN);  //  2 MB
    float*    dinv   = (float*)   alloc(sizeof(float)    * MN);
    float*    maskf  = (float*)   alloc(sizeof(float)    * MN);
    _Float16* a16    = (_Float16*)alloc(sizeof(_Float16) * MN * Dd);   //  4 MB (obs f16)
    _Float16* h16    = (_Float16*)alloc(sizeof(_Float16) * MN * Hh);   //  8 MB (h f16)
    _Float16* wp1    = (_Float16*)alloc(sizeof(_Float16) * Dd * Hh);   // 16 KB
    _Float16* wp2    = (_Float16*)alloc(sizeof(_Float16) * Hh * Hh);   // 32 KB

    const int T = 256;
    // precision packs (f32 -> f16 once; WMMA loop stays conversion-free)
    packA_kernel<<<(MN * Dd / 4 + T - 1) / T, T, 0, stream>>>(obs, a16, MN * Dd);
    packW_kernel<Dd><<<2, T, 0, stream>>>(W1, wp1);    //  512 fragments
    packW_kernel<Hh><<<4, T, 0, stream>>>(W2, wp2);    // 1024 fragments

    // graph build
    mask_kernel<<<(MN + T - 1) / T, T, 0, stream>>>(obs, maskf);
    knn_kernel<<<Bn * 4, 256, 0, stream>>>(obs, maskf, nbr);
    deg_init_kernel<<<MN / T, T, 0, stream>>>(maskf, dinv);
    deg_scatter_kernel<<<MN / T, T, 0, stream>>>(maskf, nbr, dinv);
    dinv_kernel<<<MN / T, T, 0, stream>>>(dinv);

    const int gemmBlocks = MN / 16;   // 2048 (one 16-row m-tile per block, 8 waves = 8 n-tiles)

    // layer 1: A = obs16 (MN x 64)
    gemm_wmma_kernel<Dd><<<gemmBlocks, 256, 0, stream>>>(a16, wp1, ybuf);
    zero_kernel<<<(MN * Hh) / T, T, 0, stream>>>(aggbuf, MN * Hh);
    agg_scatter_kernel<<<MN, Hh, 0, stream>>>(ybuf, dinv, maskf, nbr, aggbuf);
    combine_kernel<<<(MN * Hh) / T, T, 0, stream>>>(ybuf, aggbuf, dinv, b1, hbuf, h16);

    // layer 2: A = h16 (MN x 128)
    gemm_wmma_kernel<Hh><<<gemmBlocks, 256, 0, stream>>>(h16, wp2, ybuf);
    zero_kernel<<<(MN * Hh) / T, T, 0, stream>>>(aggbuf, MN * Hh);
    agg_scatter_kernel<<<MN, Hh, 0, stream>>>(ybuf, dinv, maskf, nbr, aggbuf);
    combine_kernel<<<(MN * Hh) / T, T, 0, stream>>>(ybuf, aggbuf, dinv, b2, hbuf, h16);

    // output head
    out_kernel<<<(MN * OUTD) / T, T, 0, stream>>>(hbuf, Wout, bout, maskf, out);
}
